// Attention_11476152615549
// MI455X (gfx1250) — compile-verified, matrix operand order checked by reading, not run on hardware
//
#include <hip/hip_runtime.h>
#include <hip/hip_bf16.h>
#include <math.h>

typedef __attribute__((ext_vector_type(16))) _Float16 v16h;
typedef __attribute__((ext_vector_type(8)))  _Float16 v8h;
typedef __attribute__((ext_vector_type(8)))  float    v8f;

#define WMMA_F16(A,B,C) __builtin_amdgcn_wmma_f32_16x16x32_f16(false,(A),false,(B),(short)0,(C),false,false)

#define NB 16    // batch
#define NC 256   // channels
#define NN 1024  // H*W
#define NH 8     // heads
#define KD 16    // key dim
#define DV 32    // value dim per head

// ---------------------------------------------------------------------------
// Kernel 1: fused QKV 1x1-conv + BN.  y[o,n] = (sum_c w[o,c] x[c,n])*s[o]+b[o]
// Output rows 0-127 -> Q, 128-255 -> K (stored [bh][n][16] f16),
// rows 256-511 -> V (stored [bh][d=32][n] f16).
// ---------------------------------------------------------------------------
__global__ __launch_bounds__(256) void qkv_kernel(
    const float* __restrict__ x,
    const float* __restrict__ wq, const float* __restrict__ sq, const float* __restrict__ bq,
    const float* __restrict__ wk, const float* __restrict__ sk, const float* __restrict__ bk,
    const float* __restrict__ wv, const float* __restrict__ sv, const float* __restrict__ bv,
    _Float16* __restrict__ Qws, _Float16* __restrict__ Kws, _Float16* __restrict__ Vws)
{
  // LDS: x tile transposed [n=64][c=256], stride 264 halves (16B-aligned rows)
  __shared__ __align__(16) _Float16 xs[64 * 264];
  const int tid = threadIdx.x;
  const int b   = blockIdx.z;
  const int n0  = blockIdx.x * 64;

  const float* xb = x + (size_t)b * NC * NN + n0;
  for (int i = tid; i < NC * 64; i += 256) {
    int c = i >> 6, n = i & 63;                 // n fastest -> coalesced global
    xs[n * 264 + c] = (_Float16)xb[(size_t)c * NN + n];
  }
  __syncthreads();

  const int wave = tid >> 5, lane = tid & 31;
  const int l16 = lane & 15, hi = lane >> 4;    // wave32: two 16-lane halves
  const int rt  = blockIdx.y * 8 + wave;        // row tile 0..31 (of 512 rows)

  const float *wm, *sm, *bm;
  int orow;
  if (rt < 8)       { wm = wq; sm = sq; bm = bq; orow = rt * 16; }
  else if (rt < 16) { wm = wk; sm = sk; bm = bk; orow = (rt - 8) * 16; }
  else              { wm = wv; sm = sv; bm = bv; orow = (rt - 16) * 16; }

  const float* arow = wm + (size_t)(orow + l16) * NC;
  v8f acc[4] = {};

#pragma unroll
  for (int ck = 0; ck < 8; ++ck) {              // K = 256 in chunks of 32
    const int cbase = ck * 32 + hi * 8;
    v16h aop;                                   // A tile 16x32 f16
    const float* ap = arow + cbase;
#pragma unroll
    for (int e = 0; e < 8; ++e) { aop[e] = (_Float16)ap[e]; aop[e + 8] = (_Float16)ap[e + 16]; }
#pragma unroll
    for (int ct = 0; ct < 4; ++ct) {            // 4 col tiles of 16
      const _Float16* bp = &xs[(ct * 16 + l16) * 264 + cbase];
      v8h blo = *(const v8h*)bp;
      v8h bhi = *(const v8h*)(bp + 16);
      v16h bop;                                 // B tile 32x16 f16
#pragma unroll
      for (int e = 0; e < 8; ++e) { bop[e] = blo[e]; bop[e + 8] = bhi[e]; }
      acc[ct] = WMMA_F16(aop, bop, acc[ct]);
    }
  }

  // epilogue: scale+bias, convert f16, layout-specific stores
#pragma unroll
  for (int ct = 0; ct < 4; ++ct) {
    const int nn = n0 + ct * 16 + l16;
    if (rt < 16) {                              // Q or K: [bh][n][16], one b128/lane
      v8h ov;
#pragma unroll
      for (int r = 0; r < 8; ++r) {
        int o = orow + r + 8 * hi;
        ov[r] = (_Float16)(acc[ct][r] * sm[o] + bm[o]);
      }
      _Float16* base = (rt < 8) ? Qws : Kws;
      _Float16* dst  = base + (((size_t)(b * NH + (rt & 7)) * NN + nn) * KD + 8 * hi);
      *(v8h*)dst = ov;
    } else {                                    // V: [bh][d][n]
      int head  = (rt - 16) >> 1;
      int dbase = ((rt - 16) & 1) * 16 + 8 * hi;
      _Float16* vdst = Vws + ((size_t)(b * NH + head) * DV + dbase) * NN + nn;
#pragma unroll
      for (int r = 0; r < 8; ++r) {
        int o = orow + r + 8 * hi;
        vdst[(size_t)r * NN] = (_Float16)(acc[ct][r] * sm[o] + bm[o]);
      }
    }
  }
}

// ---------------------------------------------------------------------------
// Kernel 2: fused flash-attention per (bh, 16-query tile), one wave each.
// Scores computed transposed (A=K-tile, B=Q-tile) so softmax reduces in-lane
// + one shfl_xor(16), and exp() results feed the PV WMMA B-operand directly.
// ---------------------------------------------------------------------------
__global__ __launch_bounds__(128) void attn_kernel(
    const _Float16* __restrict__ Qws, const _Float16* __restrict__ Kws,
    const _Float16* __restrict__ Vws, _Float16* __restrict__ Ows)
{
  const int tid  = threadIdx.x;
  const int wave = tid >> 5, lane = tid & 31;
  const int l16  = lane & 15, hi = lane >> 4;
  const int bh   = blockIdx.y;                  // 0..127
  const int qt   = blockIdx.x * 4 + wave;       // 0..63
  const int q    = qt * 16 + l16;

  // Q as B operand [kdim(32, padded) x 16 queries]; pad rows 16..31 = 0
  v16h bq = {};
  {
    const _Float16* qp = Qws + (((size_t)bh * NN + q) * KD + 8 * hi);
    v8h qv = *(const v8h*)qp;
#pragma unroll
    for (int e = 0; e < 8; ++e) bq[e] = qv[e];
  }

  v8f acc0 = {}, acc1 = {};                     // [d0..15 x q], [d16..31 x q]
  float mQ = -1e30f, lQ = 0.f;
  const v8f zc = {};
  const _Float16* kbase = Kws + (size_t)bh * NN * KD;
  const _Float16* vbase = Vws + (size_t)bh * DV * NN;

  for (int m0 = 0; m0 < NN; m0 += 32) {
    // K tiles as A operand [16 keys x kdim(32, padded)]
    v16h ka0 = {}, ka1 = {};
    {
      v8h k0 = *(const v8h*)(kbase + ((size_t)(m0 + l16) * KD + 8 * hi));
      v8h k1 = *(const v8h*)(kbase + ((size_t)(m0 + 16 + l16) * KD + 8 * hi));
#pragma unroll
      for (int e = 0; e < 8; ++e) { ka0[e] = k0[e]; ka1[e] = k1[e]; }
    }
    v8f s0 = WMMA_F16(ka0, bq, zc);             // S^T: VGPR r = key r+8*hi, lane = q
    v8f s1 = WMMA_F16(ka1, bq, zc);             // keys m0+16..m0+31

    // block max over 32 keys: 16 in-lane + complementary half via xor-16
    float bmax = -1e30f;
#pragma unroll
    for (int r = 0; r < 8; ++r) bmax = fmaxf(bmax, fmaxf(s0[r], s1[r]));
    bmax = fmaxf(bmax, __shfl_xor(bmax, 16, 32));
    float mNew = fmaxf(mQ, bmax);
    float corr = __expf(mQ - mNew);

    float psum = 0.f;
    v16h pb;                                    // P as B operand [m=32 x q=16]
#pragma unroll
    for (int r = 0; r < 8; ++r) {
      float p0 = __expf(s0[r] - mNew);
      float p1 = __expf(s1[r] - mNew);
      psum += p0 + p1;
      pb[r]     = (_Float16)p0;                 // m = r+8*hi       (matches B layout)
      pb[r + 8] = (_Float16)p1;                 // m = 16+r+8*hi
    }
    psum += __shfl_xor(psum, 16, 32);
    lQ = lQ * corr + psum;
    mQ = mNew;
#pragma unroll
    for (int r = 0; r < 8; ++r) { acc0[r] *= corr; acc1[r] *= corr; }

    // V tiles as A operand [16 d x 32 m]
    const _Float16* v0p = vbase + (size_t)l16 * NN + m0 + 8 * hi;
    const _Float16* v1p = vbase + (size_t)(16 + l16) * NN + m0 + 8 * hi;
    v8h va0l = *(const v8h*)v0p, va0h = *(const v8h*)(v0p + 16);
    v8h va1l = *(const v8h*)v1p, va1h = *(const v8h*)(v1p + 16);
    v16h va0, va1;
#pragma unroll
    for (int e = 0; e < 8; ++e) {
      va0[e] = va0l[e]; va0[e + 8] = va0h[e];
      va1[e] = va1l[e]; va1[e + 8] = va1h[e];
    }
    acc0 = WMMA_F16(va0, pb, acc0);
    acc1 = WMMA_F16(va1, pb, acc1);

    if (m0 + 32 < NN) {                         // global_prefetch_b8 for next block
      __builtin_prefetch(kbase + (size_t)(m0 + 32 + l16) * KD, 0, 0);
      __builtin_prefetch(v0p + 32, 0, 0);
      __builtin_prefetch(v1p + 32, 0, 0);
    }
  }

  // normalize, ReLU, store O as [b][h*32+d][n] f16 for the projection GEMM
  const float inv = 1.0f / lQ;
  const int b = bh >> 3, h = bh & 7;
  _Float16* obase = Ows + ((size_t)b * (NH * DV) + h * DV) * NN + q;
#pragma unroll
  for (int r = 0; r < 8; ++r) {
    int d0 = r + 8 * hi, d1 = 16 + r + 8 * hi;
    obase[(size_t)d0 * NN] = (_Float16)fmaxf(acc0[r] * inv, 0.f);
    obase[(size_t)d1 * NN] = (_Float16)fmaxf(acc1[r] * inv, 0.f);
  }
}

// ---------------------------------------------------------------------------
// Kernel 3: output projection 1x1-conv + BN, f32 result to d_out.
// ---------------------------------------------------------------------------
__global__ __launch_bounds__(256) void proj_kernel(
    const _Float16* __restrict__ Ows, const float* __restrict__ wp,
    const float* __restrict__ sp, const float* __restrict__ bp,
    float* __restrict__ out)
{
  __shared__ __align__(16) _Float16 xs[64 * 264];
  const int tid = threadIdx.x;
  const int b   = blockIdx.z;
  const int n0  = blockIdx.x * 64;

  const _Float16* ob = Ows + (size_t)b * NC * NN + n0;
  for (int i = tid; i < NC * 64; i += 256) {
    int c = i >> 6, n = i & 63;
    xs[n * 264 + c] = ob[(size_t)c * NN + n];
  }
  __syncthreads();

  const int wave = tid >> 5, lane = tid & 31;
  const int l16 = lane & 15, hi = lane >> 4;
  const int rt  = blockIdx.y * 8 + wave;        // 0..15
  const int orow = rt * 16;
  const float* arow = wp + (size_t)(orow + l16) * NC;

  v8f acc[4] = {};
#pragma unroll
  for (int ck = 0; ck < 8; ++ck) {
    const int cbase = ck * 32 + hi * 8;
    v16h aop;
    const float* ap = arow + cbase;
#pragma unroll
    for (int e = 0; e < 8; ++e) { aop[e] = (_Float16)ap[e]; aop[e + 8] = (_Float16)ap[e + 16]; }
#pragma unroll
    for (int ct = 0; ct < 4; ++ct) {
      const _Float16* bptr = &xs[(ct * 16 + l16) * 264 + cbase];
      v8h blo = *(const v8h*)bptr;
      v8h bh2 = *(const v8h*)(bptr + 16);
      v16h bop;
#pragma unroll
      for (int e = 0; e < 8; ++e) { bop[e] = blo[e]; bop[e + 8] = bh2[e]; }
      acc[ct] = WMMA_F16(aop, bop, acc[ct]);
    }
  }

#pragma unroll
  for (int ct = 0; ct < 4; ++ct) {
    const int nn = n0 + ct * 16 + l16;
    float* dst = out + (size_t)b * NC * NN + nn;
#pragma unroll
    for (int r = 0; r < 8; ++r) {
      int o = orow + r + 8 * hi;
      dst[(size_t)o * NN] = acc[ct][r] * sp[o] + bp[o];
    }
  }
}

// ---------------------------------------------------------------------------
extern "C" void kernel_launch(void* const* d_in, const int* in_sizes, int n_in,
                              void* d_out, int out_size, void* d_ws, size_t ws_size,
                              hipStream_t stream) {
  const float* x  = (const float*)d_in[0];
  const float* wq = (const float*)d_in[1];
  const float* sq = (const float*)d_in[2];
  const float* bq = (const float*)d_in[3];
  const float* wk = (const float*)d_in[4];
  const float* sk = (const float*)d_in[5];
  const float* bk = (const float*)d_in[6];
  const float* wv = (const float*)d_in[7];
  const float* sv = (const float*)d_in[8];
  const float* bv = (const float*)d_in[9];
  const float* wp = (const float*)d_in[10];
  const float* sp = (const float*)d_in[11];
  const float* bp = (const float*)d_in[12];
  float* out = (float*)d_out;

  // Workspace layout (f16): Q 4MB | K 4MB | V 8MB | O 8MB  (24MB total)
  char* wsb = (char*)d_ws;
  _Float16* Qws = (_Float16*)(wsb);
  _Float16* Kws = (_Float16*)(wsb + (4u  << 20));
  _Float16* Vws = (_Float16*)(wsb + (8u  << 20));
  _Float16* Ows = (_Float16*)(wsb + (16u << 20));

  qkv_kernel <<<dim3(16, 4, 16), 256, 0, stream>>>(x, wq, sq, bq, wk, sk, bk, wv, sv, bv, Qws, Kws, Vws);
  attn_kernel<<<dim3(16, 128),   128, 0, stream>>>(Qws, Kws, Vws, Ows);
  proj_kernel<<<dim3(16, 2, 16), 256, 0, stream>>>(Ows, wp, sp, bp, out);

  (void)in_sizes; (void)n_in; (void)out_size; (void)ws_size;
}
